// simple_GNN_AE_35691178230486
// MI455X (gfx1250) — matrix operand order repaired and checked by reading, not compiled, not run on hardware
//
#include <hip/hip_runtime.h>
#include <hip/hip_bf16.h>
#include <stdint.h>

// ---------------- problem constants (match reference) ----------------
#define NGENES 5000
#define NGRAPH 16
#define NN     (NGENES * NGRAPH)   // 80000 nodes
#define DEG    16
#define NE     (NN * DEG)          // 1,280,000 edges
#define HID    64
#define EMBD   8
#define AEH    512

typedef __attribute__((ext_vector_type(16))) _Float16 v16h;
typedef __attribute__((ext_vector_type(8)))  _Float16 v8h;
typedef __attribute__((ext_vector_type(8)))  float    v8f;

// ------------- float <-> monotonic-uint encoding for atomic max -------------
__device__ __forceinline__ unsigned fenc(float f) {
  unsigned u = __float_as_uint(f);
  return (u & 0x80000000u) ? ~u : (u | 0x80000000u);
}
__device__ __forceinline__ float fdec(unsigned u) {
  unsigned v = (u & 0x80000000u) ? (u & 0x7FFFFFFFu) : ~u;
  return __uint_as_float(v);
}
#define ENC_NEG_INF 0x007FFFFFu  // fenc(-inf)

// ---------------------------------------------------------------------------
// WMMA GEMM on pre-packed f16 operands.
//   A  : [M, K] f16 row-major      (M % 16 == 0, K % 32 == 0)
//   Bt : [Npad, K] f16 row-major   = B^T, rows n >= N are zero-padded
//   C  : [M, N] f32
// One wave per 16x16 tile. Inner loop (x2 unrolled): 8x b128 load + 2x WMMA,
// no LDS, no branches. Split-K via blockIdx.z, atomicAdd into bias-filled C.
// ---------------------------------------------------------------------------
__device__ __forceinline__ v8f wmma_step(const _Float16* __restrict__ aptr,
                                         const _Float16* __restrict__ bptr,
                                         int k0, int koff, int kb16, v8f acc) {
  v8h a0 = *(const v8h*)(aptr + k0 + koff);        // K: koff .. koff+7
  v8h a1 = *(const v8h*)(aptr + k0 + koff + 16);   // K: koff+16 .. +23
  v8h b0 = *(const v8h*)(bptr + k0 + koff);        // K: koff .. koff+7
  v8h b1 = *(const v8h*)(bptr + k0 + koff + 16);   // K: koff+16 .. +23
  v16h af = __builtin_shufflevector(a0, a1, 0,1,2,3,4,5,6,7,8,9,10,11,12,13,14,15);
  v16h bf = __builtin_shufflevector(b0, b1, 0,1,2,3,4,5,6,7,8,9,10,11,12,13,14,15);
  return __builtin_amdgcn_wmma_f32_16x16x32_f16(
      /*neg_a=*/false, af, /*neg_b=*/false, bf,
      /*c_mod=*/(short)0, acc, /*reuse_a=*/false, /*reuse_b=*/false);
}

__global__ void k_wmma_gemm_f16(const _Float16* __restrict__ A,
                                const _Float16* __restrict__ Bt,
                                const float* __restrict__ bias,
                                float* __restrict__ C,
                                int M, int N, int K, int accumulate) {
  const int lane = threadIdx.x;          // wave32
  const int m0 = blockIdx.y * 16;
  const int n0 = blockIdx.x * 16;

  const int ksteps = K >> 5;
  const int kpb    = (ksteps + gridDim.z - 1) / gridDim.z;
  const int ks0    = blockIdx.z * kpb;
  const int ks1    = (ks0 + kpb < ksteps) ? (ks0 + kpb) : ksteps;

  // CDNA5 16-bit fragment coordinates (wave32)
  const int mrow = lane & 15;            // A row within tile
  const int koff = (lane >> 4) * 8;      // A K sub-offset: lanes 16-31 -> +8
  const int col  = lane & 15;            // B column within tile
  const int kb16 = (lane >> 4) * 16;     // unused; B uses koff (+8), same as A

  const _Float16* __restrict__ aptr = A  + (long long)(m0 + mrow) * K;
  const _Float16* __restrict__ bptr = Bt + (long long)(n0 + col)  * K;

  v8f acc = {};
  int ks = ks0;
  // main loop: 2 K-steps (64 K-elements) per iteration
  for (; ks + 2 <= ks1; ks += 2) {
    const int k0 = ks << 5;
    __builtin_prefetch(aptr + k0 + 64, 0, 1);   // global_prefetch_b8, next slab
    __builtin_prefetch(bptr + k0 + 64, 0, 1);
    acc = wmma_step(aptr, bptr, k0,      koff, kb16, acc);
    acc = wmma_step(aptr, bptr, k0 + 32, koff, kb16, acc);
  }
  if (ks < ks1)                                  // odd split-K chunk tail
    acc = wmma_step(aptr, bptr, ks << 5, koff, kb16, acc);

  const int n = n0 + col;
  if (n < N) {
    const long long base = (long long)(m0 + (lane >> 4) * 8) * N + n;
    if (accumulate) {
      #pragma unroll
      for (int i = 0; i < 8; ++i) atomicAdd(&C[base + (long long)i * N], acc[i]);
    } else {
      const float badd = bias ? bias[n] : 0.0f;
      #pragma unroll
      for (int i = 0; i < 8; ++i) C[base + (long long)i * N] = acc[i] + badd;
    }
  }
}

// ------------------------- operand packing kernels -------------------------
// f32 -> f16, straight copy
__global__ void k_cvt_h(const float* __restrict__ X, _Float16* __restrict__ Y,
                        long long n) {
  long long i = (long long)blockIdx.x * blockDim.x + threadIdx.x;
  if (i < n) Y[i] = (_Float16)X[i];
}

// B [K,N] f32 -> Bt [Npad,K] f16 (transpose, zero-pad rows n >= N)
// grid: (ceil(K/256), Npad)
__global__ void k_cvt_h_t(const float* __restrict__ B, _Float16* __restrict__ Bt,
                          int K, int N) {
  int k = blockIdx.x * blockDim.x + threadIdx.x;
  int n = blockIdx.y;
  if (k >= K) return;
  float v = (n < N) ? B[(long long)k * N + n] : 0.0f;
  Bt[(long long)n * K + k] = (_Float16)v;
}

// C[m,n] = bias[n] (or 0)
__global__ void k_fill_bias(float* __restrict__ C, const float* __restrict__ bias,
                            int M, int N) {
  long long i = (long long)blockIdx.x * blockDim.x + threadIdx.x;
  if (i >= (long long)M * N) return;
  C[i] = bias ? bias[(int)(i % N)] : 0.0f;
}

// ------------------------------- GAT kernels -------------------------------
__global__ void k_att(const float* __restrict__ h, const float* __restrict__ as,
                      const float* __restrict__ ad, float* __restrict__ a_s,
                      float* __restrict__ a_d) {
  int n = blockIdx.x * blockDim.x + threadIdx.x;
  if (n >= NN) return;
  float s = 0.f, d = 0.f;
  #pragma unroll 8
  for (int j = 0; j < HID; ++j) {
    float v = h[(long long)n * HID + j];
    s += v * as[j];
    d += v * ad[j];
  }
  a_s[n] = s; a_d[n] = d;
}

__global__ void k_gat_init(unsigned* __restrict__ menc, float* __restrict__ denom) {
  int n = blockIdx.x * blockDim.x + threadIdx.x;
  if (n >= NN) return;
  menc[n]  = ENC_NEG_INF;
  denom[n] = 0.0f;
}

__global__ void k_edge_max(const long long* __restrict__ ei,
                           const float* __restrict__ a_s, const float* __restrict__ a_d,
                           unsigned* __restrict__ menc) {
  int e = blockIdx.x * blockDim.x + threadIdx.x;
  if (e >= NE + NN) return;
  int s, d;
  if (e < NE) { s = (int)ei[e]; d = (int)ei[NE + e]; }
  else        { s = d = e - NE; }
  float v = a_s[s] + a_d[d];
  v = v > 0.f ? v : 0.2f * v;
  atomicMax(&menc[d], fenc(v));
}

__global__ void k_edge_accum(const long long* __restrict__ ei,
                             const float* __restrict__ a_s, const float* __restrict__ a_d,
                             const unsigned* __restrict__ menc,
                             const float* __restrict__ h,
                             float* __restrict__ denom, float* __restrict__ agg) {
  long long t = (long long)blockIdx.x * blockDim.x + threadIdx.x;
  if (t >= (long long)(NE + NN) * HID) return;
  int f = (int)(t & (HID - 1));
  long long e = t >> 6;
  int s, d;
  if (e < NE) { s = (int)ei[e]; d = (int)ei[NE + e]; }
  else        { s = d = (int)(e - NE); }
  float v = a_s[s] + a_d[d];
  v = v > 0.f ? v : 0.2f * v;
  float w = __expf(v - fdec(menc[d]));
  if (f == 0) atomicAdd(&denom[d], w);
  atomicAdd(&agg[(long long)d * HID + f], w * h[(long long)s * HID + f]);
}

__global__ void k_finalize(const float* __restrict__ agg, const float* __restrict__ denom,
                           const float* __restrict__ bias, float* __restrict__ out) {
  long long t = (long long)blockIdx.x * blockDim.x + threadIdx.x;
  if (t >= (long long)NN * HID) return;
  int n = (int)(t >> 6), f = (int)(t & (HID - 1));
  float v = agg[t] / (denom[n] + 1e-16f) + bias[f];
  out[t] = v > 0.f ? v : 0.f;
}

// batchnorm over the 16-row batch axis + ReLU
__global__ void k_bn_relu(const float* __restrict__ X, const float* __restrict__ g,
                          const float* __restrict__ be, float* __restrict__ Y, int C) {
  int c = blockIdx.x * blockDim.x + threadIdx.x;
  if (c >= C) return;
  float mu = 0.f;
  #pragma unroll
  for (int i = 0; i < NGRAPH; ++i) mu += X[i * C + c];
  mu *= (1.0f / NGRAPH);
  float var = 0.f;
  #pragma unroll
  for (int i = 0; i < NGRAPH; ++i) { float dd = X[i * C + c] - mu; var += dd * dd; }
  var *= (1.0f / NGRAPH);
  float inv = rsqrtf(var + 1e-5f);
  float gc = g[c], bc = be[c];
  #pragma unroll
  for (int i = 0; i < NGRAPH; ++i) {
    float y = gc * (X[i * C + c] - mu) * inv + bc;
    Y[i * C + c] = y > 0.f ? y : 0.f;
  }
}

// ---------------------------------------------------------------------------
static inline dim3 lin_grid(long long n, int b) { return dim3((unsigned)((n + b - 1) / b)); }
static inline int pad16(int n) { return (n + 15) & ~15; }

struct Ctx {
  hipStream_t st;
  _Float16* ahBig;    // NN*HID halves (recycled for x / y1 / emb)
  _Float16* ahSmall;  // 16*512 halves
  _Float16* btSmall;  // up to 512*512 halves (recycled square weights)
  _Float16* btE0;     // 512*40000 halves
  _Float16* btD2;     // 5008*512 halves
};

// pack A (f32[M,K] -> f16) into dst
static void packA(Ctx& c, const float* A, _Float16* dst, long long elems) {
  k_cvt_h<<<lin_grid(elems, 256), 256, 0, c.st>>>(A, dst, elems);
}
// pack B (f32[K,N] -> f16 transposed [pad16(N),K]) into dst
static void packB(Ctx& c, const float* B, _Float16* dst, int K, int N) {
  dim3 g((K + 255) / 256, pad16(N));
  k_cvt_h_t<<<g, 256, 0, c.st>>>(B, dst, K, N);
}

static void run_gemm(Ctx& c, const _Float16* A, const _Float16* Bt, const float* bias,
                     float* C, int M, int N, int K, int splitk) {
  dim3 grid((N + 15) / 16, M / 16, splitk);
  if (splitk > 1) {
    k_fill_bias<<<lin_grid((long long)M * N, 256), 256, 0, c.st>>>(C, bias, M, N);
    k_wmma_gemm_f16<<<grid, 32, 0, c.st>>>(A, Bt, nullptr, C, M, N, K, 1);
  } else {
    k_wmma_gemm_f16<<<grid, 32, 0, c.st>>>(A, Bt, bias, C, M, N, K, 0);
  }
}

extern "C" void kernel_launch(void* const* d_in, const int* in_sizes, int n_in,
                              void* d_out, int out_size, void* d_ws, size_t ws_size,
                              hipStream_t stream) {
  (void)in_sizes; (void)n_in; (void)out_size; (void)ws_size;
  // -------- inputs (setup_inputs dict order, params lists flattened) --------
  const float*     x    = (const float*)d_in[0];
  const long long* ei   = (const long long*)d_in[1];   // int64 [2, NE]
  const float* W1  = (const float*)d_in[2];
  const float* as1 = (const float*)d_in[3];
  const float* ad1 = (const float*)d_in[4];
  const float* b1  = (const float*)d_in[5];
  const float* W2  = (const float*)d_in[6];
  const float* as2 = (const float*)d_in[7];
  const float* ad2 = (const float*)d_in[8];
  const float* b2  = (const float*)d_in[9];
  const float* Wl  = (const float*)d_in[10];
  const float* bl  = (const float*)d_in[11];
  const float* eW0 = (const float*)d_in[12]; const float* eb0 = (const float*)d_in[13];
  const float* eg0 = (const float*)d_in[14]; const float* ee0 = (const float*)d_in[15];
  const float* eW1 = (const float*)d_in[16]; const float* eb1 = (const float*)d_in[17];
  const float* eg1 = (const float*)d_in[18]; const float* ee1 = (const float*)d_in[19];
  const float* eW2 = (const float*)d_in[20]; const float* eb2 = (const float*)d_in[21];
  const float* dW0 = (const float*)d_in[22]; const float* db0 = (const float*)d_in[23];
  const float* dg0 = (const float*)d_in[24]; const float* de0 = (const float*)d_in[25];
  const float* dW1 = (const float*)d_in[26]; const float* db1 = (const float*)d_in[27];
  const float* dg1 = (const float*)d_in[28]; const float* de1 = (const float*)d_in[29];
  const float* dW2 = (const float*)d_in[30]; const float* db2 = (const float*)d_in[31];
  float* out = (float*)d_out;

  // -------- workspace layout (f32 region then f16 region) --------
  float* p = (float*)d_ws;
  float*    hbuf  = p;              p += (long long)NN * HID;    // GAT h (pre-agg)
  float*    agg   = p;              p += (long long)NN * HID;    // scatter accumulator
  float*    y1    = p;              p += (long long)NN * HID;    // layer outputs
  float*    a_s   = p;              p += NN;
  float*    a_d   = p;              p += NN;
  float*    denom = p;              p += NN;
  unsigned* menc  = (unsigned*)p;   p += NN;
  float*    emb   = p;              p += (long long)NN * EMBD;   // == [16, 40000]
  float*    zA = p; p += NGRAPH * AEH;   float* tA = p; p += NGRAPH * AEH;
  float*    zB = p; p += NGRAPH * AEH;   float* tB = p; p += NGRAPH * AEH;
  float*    zC = p; p += NGRAPH * AEH;
  float*    zD = p; p += NGRAPH * AEH;   float* tD = p; p += NGRAPH * AEH;
  float*    zE = p; p += NGRAPH * AEH;   float* tE = p; p += NGRAPH * AEH;

  Ctx c; c.st = stream;
  _Float16* hp = (_Float16*)p;
  c.ahBig   = hp;  hp += (long long)NN * HID;             // 5.12M halves
  c.ahSmall = hp;  hp += NGRAPH * AEH;
  c.btSmall = hp;  hp += (long long)AEH * AEH;            // 512x512
  c.btE0    = hp;  hp += (long long)AEH * EMBD * NGENES;  // [512][40000]
  c.btD2    = hp;  hp += (long long)pad16(NGENES) * AEH;  // [5008][512]
  _Float16* btW  = hp; hp += HID * HID;                   // [64][64] for W1/W2
  _Float16* btWl = hp; hp += 16 * HID;                    // [16][64]

  const long long EDGF = (long long)(NE + NN) * HID;

  // -------- GAT layer: yout = relu(GATConv(xin)) --------
  auto gat_layer = [&](const float* xin, const float* W, const float* as,
                       const float* ad, const float* bias, float* yout) {
    packA(c, xin, c.ahBig, (long long)NN * HID);
    packB(c, W, btW, HID, HID);
    run_gemm(c, c.ahBig, btW, nullptr, hbuf, NN, HID, HID, 1);     // h = x @ W
    k_att<<<lin_grid(NN, 256), 256, 0, stream>>>(hbuf, as, ad, a_s, a_d);
    k_gat_init<<<lin_grid(NN, 256), 256, 0, stream>>>(menc, denom);
    k_fill_bias<<<lin_grid((long long)NN * HID, 256), 256, 0, stream>>>(agg, nullptr, NN, HID);
    k_edge_max<<<lin_grid(NE + NN, 256), 256, 0, stream>>>(ei, a_s, a_d, menc);
    k_edge_accum<<<lin_grid(EDGF, 256), 256, 0, stream>>>(ei, a_s, a_d, menc, hbuf, denom, agg);
    k_finalize<<<lin_grid((long long)NN * HID, 256), 256, 0, stream>>>(agg, denom, bias, yout);
  };

  // -------- pipeline --------
  gat_layer(x,  W1, as1, ad1, b1, y1);                 // GAT 1 (+ReLU)
  gat_layer(y1, W2, as2, ad2, b2, y1);                 // GAT 2 (+ReLU)

  // embedding: [80000,64] @ [64,8] + bl -> emb ([80000,8] == [16,40000] row-major)
  packA(c, y1, c.ahBig, (long long)NN * HID);
  packB(c, Wl, btWl, HID, EMBD);
  run_gemm(c, c.ahBig, btWl, bl, emb, NN, EMBD, HID, 1);

  // encoder: 40000 -> 512 (BN+ReLU) -> 512 (BN+ReLU) -> 512
  packA(c, emb, c.ahBig, (long long)NGRAPH * EMBD * NGENES);
  packB(c, eW0, c.btE0, EMBD * NGENES, AEH);
  run_gemm(c, c.ahBig, c.btE0, eb0, zA, NGRAPH, AEH, EMBD * NGENES, 100);
  k_bn_relu<<<lin_grid(AEH, 256), 256, 0, stream>>>(zA, eg0, ee0, tA, AEH);

  packA(c, tA, c.ahSmall, NGRAPH * AEH);
  packB(c, eW1, c.btSmall, AEH, AEH);
  run_gemm(c, c.ahSmall, c.btSmall, eb1, zB, NGRAPH, AEH, AEH, 4);
  k_bn_relu<<<lin_grid(AEH, 256), 256, 0, stream>>>(zB, eg1, ee1, tB, AEH);

  packA(c, tB, c.ahSmall, NGRAPH * AEH);
  packB(c, eW2, c.btSmall, AEH, AEH);
  run_gemm(c, c.ahSmall, c.btSmall, eb2, zC, NGRAPH, AEH, AEH, 4);

  // decoder: 512 -> 512 (BN+ReLU) -> 512 (BN+ReLU) -> 5000
  packA(c, zC, c.ahSmall, NGRAPH * AEH);
  packB(c, dW0, c.btSmall, AEH, AEH);
  run_gemm(c, c.ahSmall, c.btSmall, db0, zD, NGRAPH, AEH, AEH, 4);
  k_bn_relu<<<lin_grid(AEH, 256), 256, 0, stream>>>(zD, dg0, de0, tD, AEH);

  packA(c, tD, c.ahSmall, NGRAPH * AEH);
  packB(c, dW1, c.btSmall, AEH, AEH);
  run_gemm(c, c.ahSmall, c.btSmall, db1, zE, NGRAPH, AEH, AEH, 4);
  k_bn_relu<<<lin_grid(AEH, 256), 256, 0, stream>>>(zE, dg1, de1, tE, AEH);

  packA(c, tE, c.ahSmall, NGRAPH * AEH);
  packB(c, dW2, c.btD2, AEH, NGENES);
  run_gemm(c, c.ahSmall, c.btD2, db2, out, NGRAPH, NGENES, AEH, 2);  // [16,5000]
}
